// EN_CAM_75359496175780
// MI455X (gfx1250) — matrix operand order repaired and checked by reading, not compiled, not run on hardware
//
#include <hip/hip_runtime.h>

typedef __attribute__((ext_vector_type(16))) _Float16 v16h;
typedef __attribute__((ext_vector_type(8)))  _Float16 v8h;
typedef __attribute__((ext_vector_type(8)))  float    v8f;
typedef __attribute__((ext_vector_type(4)))  unsigned int v4u;
typedef __attribute__((ext_vector_type(8)))  int      v8i;
typedef __attribute__((ext_vector_type(4)))  int      v4i;

static constexpr int kB   = 4;
static constexpr int kN   = 4096;   // 64*64 tokens after space_to_depth
static constexpr int kD   = 32;     // q/k dim
static constexpr int kDV  = 256;    // v dim
static constexpr int kPSTR = 48;    // LDS row stride (f16) for P tile, 16B aligned

// v_permlane16_b32 based xor-shuffle within each 16-lane row (VALU, no LDS).
__device__ __forceinline__ float permlane16_f(float v, int s1, int s2) {
    int i = __builtin_bit_cast(int, v);
    i = __builtin_amdgcn_permlane16(i, i, s1, s2, false, false);
    return __builtin_bit_cast(float, i);
}
__device__ __forceinline__ float rowmax16(float v) {
    v = fmaxf(v, permlane16_f(v, 0x67452301, 0xEFCDAB89));  // xor 1
    v = fmaxf(v, permlane16_f(v, 0x54761032, 0xDCFE98BA));  // xor 2
    v = fmaxf(v, permlane16_f(v, 0x32107654, 0xBA98FEDC));  // xor 4
    v = fmaxf(v, permlane16_f(v, 0xFEDCBA98, 0x76543210));  // xor 8
    return v;
}

// TDM: stage a [256 x 32] f16 tile of Vt (row stride kN elems) into LDS.
// D# fields per CDNA5 ISA 8.3-8.5: count=1, type=2(image), data_size=1(2B),
// tensor_dim0=kN, tensor_dim1=kDV, tile_dim0=32, tile_dim1=kDV, dim0_stride=kN.
__device__ __forceinline__ void tdm_load_v(const _Float16* gptr, unsigned lds_off) {
    unsigned long long ga = (unsigned long long)(uintptr_t)gptr;
    v4u g0;
    g0[0] = 1u;                                          // count=1 (valid), user mode
    g0[1] = lds_off;                                     // LDS byte address
    g0[2] = (unsigned)(ga & 0xffffffffu);                // global_addr[31:0]
    g0[3] = (unsigned)((ga >> 32) & 0x01ffffffu) | 0x80000000u; // addr[56:32]|type=2
    v8i g1;
    g1[0] = 0x00010000;                                  // wg_mask=0, data_size=1 (2B)
    g1[1] = (int)((unsigned)kN << 16);                   // tensor_dim0[15:0] in [63:48]
    g1[2] = (int)(((unsigned)kN >> 16) | ((unsigned)kDV << 16)); // dim0 hi | dim1 lo
    g1[3] = (int)(((unsigned)kDV >> 16) | (32u << 16));  // dim1 hi | tile_dim0=32
    g1[4] = kDV;                                         // tile_dim1=256, tile_dim2=0
    g1[5] = kN;                                          // tensor_dim0_stride[31:0]
    g1[6] = 0;                                           // stride hi, dim1_stride lo
    g1[7] = 0;
    v4i g2 = {0, 0, 0, 0};
    v4i g3 = {0, 0, 0, 0};
    v8i g4 = {0, 0, 0, 0, 0, 0, 0, 0};                   // 6-arg toolchain variant
    __builtin_amdgcn_tensor_load_to_lds(g0, g1, g2, g3, g4, 0);
}

// -------------------------------------------------------------------------
// Kernel 1: fused space_to_depth gather + 1x1-conv projections.
// Produces Qh,Kh as [b][n][32] f16 and Vt as [b][c][n] f16.
// -------------------------------------------------------------------------
__global__ __launch_bounds__(320) void proj_kernel(
    const float* __restrict__ x,
    const float* __restrict__ wq, const float* __restrict__ bq,
    const float* __restrict__ wk, const float* __restrict__ bk,
    const float* __restrict__ wv, const float* __restrict__ bvp,
    _Float16* __restrict__ Qh, _Float16* __restrict__ Kh,
    _Float16* __restrict__ Vt)
{
    __shared__ float xs[256 * 16];               // xp for 16 pixels
    const int b  = blockIdx.x >> 8;              // 256 pixel-groups / batch
    const int n0 = (blockIdx.x & 255) << 4;

    for (int t = threadIdx.x; t < 256 * 16; t += 320) {
        int c = t >> 4, px = t & 15;
        int n  = n0 + px;
        int h2 = n >> 6, w2 = n & 63;
        int cc = c & 63, hb = (c >> 6) & 1, wb = (c >> 7) & 1;
        size_t idx = (((size_t)b * 64 + cc) * 128 + (h2 * 2 + hb)) * 128
                   + (w2 * 2 + wb);
        xs[t] = x[idx];
    }
    __syncthreads();

    const int o = threadIdx.x;                   // 0..319 output channel
    const float* wrow; float bias;
    if (o < 32)      { wrow = wq + o * 256;        bias = bq[o];        }
    else if (o < 64) { wrow = wk + (o - 32) * 256; bias = bk[o - 32];   }
    else             { wrow = wv + (o - 64) * 256; bias = bvp[o - 64];  }

    float acc[16];
    #pragma unroll
    for (int px = 0; px < 16; ++px) acc[px] = bias;

    for (int c = 0; c < 256; ++c) {
        float w = wrow[c];
        const float* xr = &xs[c * 16];
        #pragma unroll
        for (int px = 0; px < 16; ++px) acc[px] = fmaf(w, xr[px], acc[px]);
    }

    if (o < 32) {
        _Float16* qd = Qh + ((size_t)b * kN + n0) * kD + o;
        #pragma unroll
        for (int px = 0; px < 16; ++px) qd[(size_t)px * kD] = (_Float16)acc[px];
    } else if (o < 64) {
        _Float16* kd = Kh + ((size_t)b * kN + n0) * kD + (o - 32);
        #pragma unroll
        for (int px = 0; px < 16; ++px) kd[(size_t)px * kD] = (_Float16)acc[px];
    } else {
        _Float16* vd = Vt + ((size_t)b * kDV + (o - 64)) * kN + n0;
        #pragma unroll
        for (int px = 0; px < 16; ++px) vd[px] = (_Float16)acc[px];
    }
}

// -------------------------------------------------------------------------
// Kernel 2: flash attention, 8 waves / WG (8 query tiles, same batch).
// V tiles staged once per WG by the Tensor Data Mover into double-buffered
// LDS (8x cut in L2->WGP traffic); per-wave: 2 QK^T WMMAs, permlane16 row-max,
// WMMA-vs-ones row-sum, 16 PV WMMAs from LDS V fragments.
// -------------------------------------------------------------------------
__global__ __launch_bounds__(256) void attn_kernel(
    const _Float16* __restrict__ Qh, const _Float16* __restrict__ Kh,
    const _Float16* __restrict__ Vt, const float* __restrict__ x,
    const float* __restrict__ gamma, float* __restrict__ out)
{
    __shared__ __align__(128) _Float16 vbuf[2][kDV * 32];   // 2 x 16 KB V tiles
    __shared__ __align__(128) _Float16 lds_p[8 * 16 * kPSTR]; // 12 KB P staging

    const int wid  = threadIdx.x >> 5;
    const int lane = threadIdx.x & 31;
    const int hl   = lane >> 4;
    const int l15  = lane & 15;
    const int qt   = blockIdx.x * 8 + wid;       // 8 query tiles per WG
    const int b    = qt >> 8;                    // same b for all waves in WG
    const int n0   = (qt & 255) << 4;
    _Float16* pbuf = lds_p + wid * 16 * kPSTR;

    const unsigned vb_off0 = (unsigned)(uintptr_t)(&vbuf[0][0]);
    const unsigned vb_off1 = (unsigned)(uintptr_t)(&vbuf[1][0]);
    const _Float16* vglob  = Vt + (size_t)b * kDV * kN;

    // A-fragment of Q (16x32 f16)
    const _Float16* qr = Qh + ((size_t)b * kN + n0 + l15) * kD;
    v8h qa = *(const v8h*)(qr + hl * 16);
    v8h qb = *(const v8h*)(qr + hl * 16 + 8);
    v16h aq = __builtin_shufflevector(qa, qb, 0,1,2,3,4,5,6,7,8,9,10,11,12,13,14,15);

    v8f acc[16];
    v8f acc_l;                                   // running softmax denominator
    float m_run[8];
    #pragma unroll
    for (int t = 0; t < 16; ++t)
        #pragma unroll
        for (int r = 0; r < 8; ++r) acc[t][r] = 0.0f;
    #pragma unroll
    for (int r = 0; r < 8; ++r) { acc_l[r] = 0.0f; m_run[r] = -3.0e38f; }

    const v8f zero8 = {0.f,0.f,0.f,0.f,0.f,0.f,0.f,0.f};
    v16h vones;
    #pragma unroll
    for (int i = 0; i < 16; ++i) vones[i] = (_Float16)1.0f;

    // TDM prologue: stage V block 0 into buffer 0
    if (wid == 0) tdm_load_v(vglob, vb_off0);
    int cur = 0;

    for (int kb = 0; kb < kN; kb += 32) {
        if (wid == 0) __builtin_amdgcn_s_wait_tensorcnt(0);  // cur buffer ready
        __syncthreads();                                     // publish to all waves
        if (wid == 0 && kb + 32 < kN)                        // prefetch next V tile
            tdm_load_v(vglob + kb + 32, cur ? vb_off0 : vb_off1);

        // K^T B-fragments straight from global (K panel is small, L2-resident)
        v16h bk0 = *(const v16h*)(Kh + ((size_t)b * kN + kb + l15) * kD + hl * 16);
        v16h bk1 = *(const v16h*)(Kh + ((size_t)b * kN + kb + 16 + l15) * kD + hl * 16);
        if (kb + 32 < kN)
            __builtin_prefetch(Kh + ((size_t)b * kN + kb + 32 + l15) * kD, 0, 3);

        v8f s0 = __builtin_amdgcn_wmma_f32_16x16x32_f16(false, aq, false, bk0,
                                                        (short)0, zero8, false, false);
        v8f s1 = __builtin_amdgcn_wmma_f32_16x16x32_f16(false, aq, false, bk1,
                                                        (short)0, zero8, false, false);

        // online softmax: row-max via permlane16 butterflies (pure VALU)
        float sc[8];
        #pragma unroll
        for (int r = 0; r < 8; ++r) {
            float mb = rowmax16(fmaxf(s0[r], s1[r]));
            float mn = fmaxf(m_run[r], mb);
            sc[r] = __expf(m_run[r] - mn);
            m_run[r] = mn;
            float p0 = __expf(s0[r] - mn);
            float p1 = __expf(s1[r] - mn);
            pbuf[(r + hl * 8) * kPSTR + l15]      = (_Float16)p0;
            pbuf[(r + hl * 8) * kPSTR + 16 + l15] = (_Float16)p1;
        }
        #pragma unroll
        for (int r = 0; r < 8; ++r) acc_l[r] *= sc[r];
        #pragma unroll
        for (int t = 0; t < 16; ++t)
            #pragma unroll
            for (int r = 0; r < 8; ++r) acc[t][r] *= sc[r];

        asm volatile("s_wait_dscnt 0" ::: "memory");   // wave-local P RAW fence

        // A-fragment of P (16x32 f16) from LDS
        const v8h* pr = (const v8h*)(pbuf + l15 * kPSTR);
        v8h pa = pr[2 * hl];
        v8h pb = pr[2 * hl + 1];
        v16h ap = __builtin_shufflevector(pa, pb, 0,1,2,3,4,5,6,7,8,9,10,11,12,13,14,15);

        // denominator: rowsum(P) via WMMA against all-ones B
        acc_l = __builtin_amdgcn_wmma_f32_16x16x32_f16(false, ap, false, vones,
                                                       (short)0, acc_l, false, false);

        // O += P * V over 16 channel tiles; V fragments from TDM-staged LDS tile
        const _Float16* vt = &vbuf[cur][0];
        #pragma unroll
        for (int t = 0; t < 16; ++t) {
            v16h bvf = *(const v16h*)(vt + (t * 16 + l15) * 32 + hl * 16);
            acc[t] = __builtin_amdgcn_wmma_f32_16x16x32_f16(false, ap, false, bvf,
                                                            (short)0, acc[t], false, false);
        }
        cur ^= 1;
    }

    // epilogue: out = gamma * (acc / rowsum) + xp, scattered via depth_to_space
    float g = gamma[0];
    float rinv[8];
    #pragma unroll
    for (int r = 0; r < 8; ++r) rinv[r] = g / acc_l[r];

    #pragma unroll
    for (int t = 0; t < 16; ++t) {
        int c  = t * 16 + l15;
        int cc = c & 63, hb = (c >> 6) & 1, wb = (c >> 7) & 1;
        #pragma unroll
        for (int r = 0; r < 8; ++r) {
            int n  = n0 + r + hl * 8;
            int h2 = n >> 6, w2 = n & 63;
            size_t idx = (((size_t)b * 64 + cc) * 128 + (h2 * 2 + hb)) * 128
                       + (w2 * 2 + wb);
            out[idx] = fmaf(acc[t][r], rinv[r], x[idx]);
        }
    }
}

// -------------------------------------------------------------------------
extern "C" void kernel_launch(void* const* d_in, const int* in_sizes, int n_in,
                              void* d_out, int out_size, void* d_ws, size_t ws_size,
                              hipStream_t stream) {
    const float* x     = (const float*)d_in[0];
    const float* wq    = (const float*)d_in[1];
    const float* bq    = (const float*)d_in[2];
    const float* wk    = (const float*)d_in[3];
    const float* bk    = (const float*)d_in[4];
    const float* wv    = (const float*)d_in[5];
    const float* bv    = (const float*)d_in[6];
    const float* gamma = (const float*)d_in[7];

    _Float16* Qh = (_Float16*)d_ws;                       // 4*4096*32  f16 = 1 MB
    _Float16* Kh = Qh + (size_t)kB * kN * kD;             // 1 MB
    _Float16* Vt = Kh + (size_t)kB * kN * kD;             // 8 MB

    proj_kernel<<<dim3(kB * (kN / 16)), dim3(320), 0, stream>>>(
        x, wq, bq, wk, bk, wv, bv, Qh, Kh, Vt);

    attn_kernel<<<dim3(kB * (kN / 16) / 8), dim3(256), 0, stream>>>(
        Qh, Kh, Vt, x, gamma, (float*)d_out);
}
